// EdgeConvBlock_74603581932065
// MI455X (gfx1250) — compile-verified, matrix operand order checked by reading, not run on hardware
//
#include <hip/hip_runtime.h>

#define N_NODES 20000
#define N_EDGES 320000
#define IN_F    128
#define H_F     256
#define BN_EPS  1e-5f
#define SELU_L  1.0507009873554805f
#define SELU_LA 1.7580993408473766f   // lambda * alpha

typedef __bf16 v8bf  __attribute__((ext_vector_type(8)));
typedef __bf16 v16bf __attribute__((ext_vector_type(16)));
typedef float  v8f   __attribute__((ext_vector_type(8)));

union ABu { v16bf v; v8bf h[2]; };

// ---------------------------------------------------------------------------
__global__ __launch_bounds__(256) void zero_ws_k(float* __restrict__ agg,
                                                 float* __restrict__ cnt) {
  int i = blockIdx.x * 256 + threadIdx.x;
  if (i < N_NODES * H_F) agg[i] = 0.f;
  if (i < N_NODES)       cnt[i] = 0.f;
}

// ---------------------------------------------------------------------------
// Fold BN scale into weights, convert to bf16, pre-pack into the WMMA
// B-fragment layout: flat = ((layer*16+nt)*8+kt)*512 + lane*16 + t
// lane<16 holds column n=lane, K slots {0..7,16..23}; lane>=16 holds n=lane-16,
// K slots {8..15,24..31} (mirrors the 16-bit A layout, transposed).
__global__ __launch_bounds__(256) void prep_w_k(const float* __restrict__ W0,
                                                const float* __restrict__ g0,
                                                const float* __restrict__ v0,
                                                const float* __restrict__ Ws,
                                                const float* __restrict__ gs,
                                                const float* __restrict__ vs,
                                                __bf16* __restrict__ Wp) {
  int idx = blockIdx.x * 256 + threadIdx.x;
  if (idx >= 3 * 65536) return;
  int t     = idx & 15;
  int lane  = (idx >> 4) & 31;
  int kt    = (idx >> 9) & 7;
  int nt    = (idx >> 12) & 15;
  int layer = idx >> 16;
  int half  = lane >> 4;
  int n     = nt * 16 + (lane & 15);
  int k     = kt * 32 + ((t < 8) ? t : t + 8) + half * 8;
  float w, g, v;
  if (layer == 0) { w = W0[n * 256 + k]; g = g0[n]; v = v0[n]; }
  else {
    int o = (layer - 1) * 256 + n;
    w = Ws[(layer - 1) * 65536 + n * 256 + k]; g = gs[o]; v = vs[o];
  }
  float s = g * rsqrtf(v + BN_EPS);
  Wp[idx] = (__bf16)(w * s);
}

// Folded bias: b' = s*(b-m) + beta  (3 layers x 256)
__global__ __launch_bounds__(256) void prep_b_k(const float* b0, const float* g0,
                                                const float* be0, const float* m0,
                                                const float* v0,
                                                const float* bs, const float* gs,
                                                const float* bes, const float* ms,
                                                const float* vs,
                                                float* __restrict__ bp) {
  int idx = blockIdx.x * 256 + threadIdx.x;
  if (idx >= 3 * 256) return;
  int layer = idx >> 8, n = idx & 255;
  float b, g, be, m, v;
  if (layer == 0) { b = b0[n]; g = g0[n]; be = be0[n]; m = m0[n]; v = v0[n]; }
  else {
    int o = (layer - 1) * 256 + n;
    b = bs[o]; g = gs[o]; be = bes[o]; m = ms[o]; v = vs[o];
  }
  float s = g * rsqrtf(v + BN_EPS);
  bp[idx] = s * (b - m) + be;
}

// ---------------------------------------------------------------------------
__device__ __forceinline__ void load_bfrag(ABu& b, const __bf16* wp) {
  b.h[0] = *(const v8bf*)wp;
  b.h[1] = *(const v8bf*)(wp + 8);
}

__device__ __forceinline__ void load_afrag(ABu* a, const __bf16* bufc, int kt,
                                           int mrow, int half) {
  const int c0 = kt * 4 + half;       // K slots {0..7}+half*8
  const int c1 = c0 + 2;              // K slots {16..23}+half*8
  #pragma unroll
  for (int mt = 0; mt < 4; ++mt) {
    const int m  = mt * 16 + mrow;
    const int sw = m & 15;
    a[mt].h[0] = *(const v8bf*)&bufc[m * 256 + ((c0 ^ sw) << 3)];
    a[mt].h[1] = *(const v8bf*)&bufc[m * 256 + ((c1 ^ sw) << 3)];
  }
}

// Fused edge MLP: 64 edges per workgroup, 8 waves, 3 fused bf16 WMMA GEMM
// layers out of a ping-pong LDS buffer. XOR-swizzled 16B chunks keep the
// stride-512B A-fragment ds_load_b128s conflict-free. Fragment loads are
// software-pipelined (double-buffered) one K-step ahead of the WMMAs.
__global__ __launch_bounds__(256) void edge_mlp_k(const float* __restrict__ x,
                                                  const int*   __restrict__ ei,
                                                  const __bf16* __restrict__ Wp,
                                                  const float* __restrict__ bp,
                                                  float* __restrict__ h_out,
                                                  float* __restrict__ agg,
                                                  float* __restrict__ cnt) {
  __shared__ __bf16 buf[2][64 * 256];   // exactly 64 KB
  const int base = blockIdx.x * 64;
  const int tid  = threadIdx.x;
  const int lane = tid & 31;
  const int wave = tid >> 5;

  // one count per valid edge
  if (tid < 64) {
    int e = base + tid;
    if (e < N_EDGES) atomicAdd(&cnt[ei[N_EDGES + e]], 1.0f);
  }

  // gather + concat + fp32->bf16 into buf[0]
  #pragma unroll
  for (int i = 0; i < 8; ++i) {
    int task = tid + i * 256;           // 64 rows x 32 chunks
    int m = task >> 5, c = task & 31;
    int e = base + m; if (e >= N_EDGES) e = N_EDGES - 1;
    int node = (c < 16) ? ei[N_EDGES + e] : ei[e];  // dst feats first, then src
    int f0 = (c & 15) * 8;
    const float4* p = (const float4*)(x + (size_t)node * IN_F + f0);
    float4 A = p[0], B = p[1];
    v8bf vv;
    vv[0] = (__bf16)A.x; vv[1] = (__bf16)A.y; vv[2] = (__bf16)A.z; vv[3] = (__bf16)A.w;
    vv[4] = (__bf16)B.x; vv[5] = (__bf16)B.y; vv[6] = (__bf16)B.z; vv[7] = (__bf16)B.w;
    *(v8bf*)&buf[0][m * 256 + ((c ^ (m & 15)) << 3)] = vv;
  }
  __syncthreads();

  int cur = 0;
  const int mrow = lane & 15;
  const int half = lane >> 4;

  for (int layer = 0; layer < 3; ++layer) {
    const __bf16* W    = Wp + layer * (16 * 8 * 512);
    const float*  bias = bp + layer * 256;
    const __bf16* bufc = &buf[cur][0];

    for (int nt2 = 0; nt2 < 2; ++nt2) {
      const int nt = wave * 2 + nt2;
      v8f z = {0.f, 0.f, 0.f, 0.f, 0.f, 0.f, 0.f, 0.f};
      v8f acc0 = z, acc1 = z, acc2 = z, acc3 = z;

      const __bf16* wbase = W + ((nt * 8) << 9) + lane * 16;

      // prologue: kt = 0 fragments
      ABu bcur; ABu acur[4];
      load_bfrag(bcur, wbase);
      load_afrag(acur, bufc, 0, mrow, half);

      #pragma unroll
      for (int kt = 0; kt < 8; ++kt) {
        ABu bnxt; ABu anxt[4];
        if (kt < 7) {                       // prefetch next K-step
          load_bfrag(bnxt, wbase + ((kt + 1) << 9));
          load_afrag(anxt, bufc, kt + 1, mrow, half);
        }
        acc0 = __builtin_amdgcn_wmma_f32_16x16x32_bf16(
            false, acur[0].v, false, bcur.v, (short)0, acc0, false, false);
        acc1 = __builtin_amdgcn_wmma_f32_16x16x32_bf16(
            false, acur[1].v, false, bcur.v, (short)0, acc1, false, false);
        acc2 = __builtin_amdgcn_wmma_f32_16x16x32_bf16(
            false, acur[2].v, false, bcur.v, (short)0, acc2, false, false);
        acc3 = __builtin_amdgcn_wmma_f32_16x16x32_bf16(
            false, acur[3].v, false, bcur.v, (short)0, acc3, false, false);
        if (kt < 7) {
          bcur = bnxt;
          #pragma unroll
          for (int mt = 0; mt < 4; ++mt) acur[mt] = anxt[mt];
        }
      }

      // epilogue: bias + SELU, then LDS (layers 0,1) or global + atomics (layer 2)
      const int   ncol = nt * 16 + (lane & 15);
      const float bv   = bias[ncol];
      #pragma unroll
      for (int mt = 0; mt < 4; ++mt) {
        v8f accv = (mt == 0) ? acc0 : (mt == 1) ? acc1 : (mt == 2) ? acc2 : acc3;
        #pragma unroll
        for (int vv = 0; vv < 8; ++vv) {
          float y = accv[vv] + bv;
          y = (y > 0.f) ? SELU_L * y : SELU_LA * (__expf(y) - 1.f);
          const int m = mt * 16 + half * 8 + vv;   // C layout: VGPR v -> row v + 8*half
          if (layer < 2) {
            const int sw = m & 15;
            const int c  = ncol >> 3;
            buf[cur ^ 1][m * 256 + ((c ^ sw) << 3) + (ncol & 7)] = (__bf16)y;
          } else {
            const int e = base + m;
            if (e < N_EDGES) {
              h_out[(size_t)e * H_F + ncol] = y;
              const int d = ei[N_EDGES + e];
              atomicAdd(&agg[d * H_F + ncol], y);
            }
          }
        }
      }
    }
    if (layer < 2) { __syncthreads(); cur ^= 1; }
  }
}

// ---------------------------------------------------------------------------
__global__ __launch_bounds__(256) void finalize_k(const float* __restrict__ agg,
                                                  const float* __restrict__ cnt,
                                                  float* __restrict__ out) {
  int i = blockIdx.x * 256 + threadIdx.x;
  if (i >= N_NODES * H_F) return;
  out[i] = agg[i] / fmaxf(cnt[i >> 8], 1.f);
}

// ---------------------------------------------------------------------------
extern "C" void kernel_launch(void* const* d_in, const int* in_sizes, int n_in,
                              void* d_out, int out_size, void* d_ws, size_t ws_size,
                              hipStream_t stream) {
  const float* x   = (const float*)d_in[0];
  const int*   ei  = (const int*)  d_in[1];
  const float* W0  = (const float*)d_in[2];
  const float* b0  = (const float*)d_in[3];
  const float* g0  = (const float*)d_in[4];
  const float* be0 = (const float*)d_in[5];
  const float* m0  = (const float*)d_in[6];
  const float* v0  = (const float*)d_in[7];
  const float* Ws  = (const float*)d_in[8];
  const float* bs  = (const float*)d_in[9];
  const float* gs  = (const float*)d_in[10];
  const float* bes = (const float*)d_in[11];
  const float* ms  = (const float*)d_in[12];
  const float* vs  = (const float*)d_in[13];

  float* out_agg = (float*)d_out;                       // [N, 256]
  float* out_h   = out_agg + (size_t)N_NODES * H_F;     // [E, 256]

  // workspace layout
  float*  agg = (float*)d_ws;                 // N*256
  float*  cnt = agg + (size_t)N_NODES * H_F;  // N
  float*  bp  = cnt + N_NODES;                // 3*256
  __bf16* Wp  = (__bf16*)(bp + 3 * 256);      // 3*65536 bf16 (16B aligned)

  zero_ws_k<<<(N_NODES * H_F + 255) / 256, 256, 0, stream>>>(agg, cnt);
  prep_w_k<<<(3 * 65536 + 255) / 256, 256, 0, stream>>>(W0, g0, v0, Ws, gs, vs, Wp);
  prep_b_k<<<3, 256, 0, stream>>>(b0, g0, be0, m0, v0, bs, gs, bes, ms, vs, bp);
  edge_mlp_k<<<(N_EDGES + 63) / 64, 256, 0, stream>>>(x, ei, Wp, bp, out_h, agg, cnt);
  finalize_k<<<(N_NODES * H_F + 255) / 256, 256, 0, stream>>>(agg, cnt, out_agg);
}
